// FlaxLLaMAAttention_69080253989571
// MI455X (gfx1250) — matrix operand order repaired and checked by reading, not compiled, not run on hardware
//
#include <hip/hip_runtime.h>
#include <cstdint>

#define DEV_INLINE __device__ __forceinline__

typedef __attribute__((ext_vector_type(16))) __bf16 bf16x16;
typedef __attribute__((ext_vector_type(8)))  __bf16 bf16x8;
typedef __attribute__((ext_vector_type(8)))  float  f32x8;
typedef __attribute__((ext_vector_type(4)))  unsigned int u32x4;
typedef __attribute__((ext_vector_type(8)))  unsigned int u32x8;

namespace cfg {
constexpr int B  = 2;
constexpr int S  = 2048;
constexpr int D  = 4096;
constexpr int H  = 32;
constexpr int HD = 128;
constexpr int M  = B * S;                 // 4096 token rows
constexpr float SM_SCALE = 0.08838834764831845f;  // 1/sqrt(128)
}

// ---------------------------------------------------------------------------
// Fragment load helper: returns 16 bf16 built from two 8-element (16B) chunks.
// A-frag (16x32, 16-bit): chunks at kbase and kbase+16  -> ld_frag16(p, p+16)
// B-frag (32x16, 16-bit): chunks at kbase and kbase+16  -> ld_frag16(p, p+16)
// ---------------------------------------------------------------------------
DEV_INLINE bf16x16 ld_frag16(const __bf16* lo_p, const __bf16* hi_p) {
  bf16x8 lo = *(const bf16x8*)lo_p;
  bf16x8 hi = *(const bf16x8*)hi_p;
  return __builtin_shufflevector(lo, hi, 0, 1, 2, 3, 4, 5, 6, 7,
                                         8, 9, 10, 11, 12, 13, 14, 15);
}

DEV_INLINE f32x8 wmma_bf16(bf16x16 a, bf16x16 b, f32x8 c) {
  return __builtin_amdgcn_wmma_f32_16x16x32_bf16(
      /*neg_a=*/false, a, /*neg_b=*/false, b,
      /*c_mod=*/(short)0, c, /*reuse_a=*/false, /*reuse_b=*/false);
}

// Low 32 bits of a generic pointer to LDS == byte offset within LDS
// (aperture check uses addr[63:32]; the LDS address is the low 32 bits, per ISA 10.2).
DEV_INLINE uint32_t lds_byte_off(const void* p) {
  return (uint32_t)(uintptr_t)p;
}

// ---------------------------------------------------------------------------
// TDM: issue a 2D tensor-tile DMA load (global -> LDS). Scalar op, TENSORcnt.
// tile_w elements per row (contiguous), tile_h rows, row stride `stride`
// elements; LDS destination padded by (pad_amt+1) DWORDs every
// 2^(pad_int+1) DWORDs. D# layout per CDNA5 ISA.
// All arguments must be wave-uniform (descriptor lives in SGPRs).
// ---------------------------------------------------------------------------
DEV_INLINE void tdm_load_2d(const void* gsrc, uint32_t lds_off,
                            uint32_t tile_w, uint32_t tile_h,
                            uint32_t stride, uint32_t td0, uint32_t td1,
                            uint32_t pad_int, uint32_t pad_amt) {
  const uint64_t ga = (uint64_t)(uintptr_t)gsrc;
  u32x4 g0;
  g0.x = 1u;                                            // count=1 (valid D#)
  g0.y = lds_off;                                       // lds_addr
  g0.z = (uint32_t)ga;                                  // global_addr[31:0]
  g0.w = (uint32_t)((ga >> 32) & 0x01FFFFFFu)           // global_addr[56:32]
         | (2u << 30);                                  // type=2 ("image")
  u32x8 g1;
  g1.s0 = (1u << 16)                 // data_size=1 -> 2 bytes
        | (1u << 20)                 // pad_enable
        | (pad_int << 22)            // pad every 2^(c+1) DWORDs
        | (pad_amt << 25);           // pad (a+1) DWORDs
  g1.s1 = td0 << 16;                 // tensor_dim0[15:0]   (bits 63:48)
  g1.s2 = (td0 >> 16) | (td1 << 16); // tensor_dim0[31:16], tensor_dim1[15:0]
  g1.s3 = (td1 >> 16) | (tile_w << 16); // tensor_dim1[31:16], tile_dim0
  g1.s4 = tile_h;                    // tile_dim1, tile_dim2=0
  g1.s5 = stride;                    // tensor_dim0_stride[31:0]
  g1.s6 = 0u;
  g1.s7 = 0u;
  asm volatile("tensor_load_to_lds %0, %1" :: "s"(g0), "s"(g1) : "memory");
}

// ---------------------------------------------------------------------------
// fp32 -> bf16 elementwise convert
// ---------------------------------------------------------------------------
__global__ __launch_bounds__(256)
void convert_f32_to_bf16(const float* __restrict__ in, __bf16* __restrict__ out,
                         long n) {
  long i = (long)blockIdx.x * blockDim.x + threadIdx.x;
  long stride = (long)gridDim.x * blockDim.x;
  for (; i < n; i += stride) out[i] = (__bf16)in[i];
}

// ---------------------------------------------------------------------------
// Transpose + convert: W[k][n] fp32 (KxN row-major)  ->  Wt[n][k] bf16
// 32x32 LDS tiles, block = 256 threads (32x8 logical).
// ---------------------------------------------------------------------------
__global__ __launch_bounds__(256)
void transpose_convert_bf16(const float* __restrict__ W, __bf16* __restrict__ Wt,
                            int K, int N) {
  __shared__ float tile[32][33];
  const int tx = threadIdx.x & 31;
  const int ty = threadIdx.x >> 5;     // 0..7
  const int n0 = blockIdx.x * 32;
  const int k0 = blockIdx.y * 32;
#pragma unroll
  for (int i = 0; i < 4; ++i)
    tile[ty + i * 8][tx] = W[(long)(k0 + ty + i * 8) * N + n0 + tx];
  __syncthreads();
#pragma unroll
  for (int i = 0; i < 4; ++i)
    Wt[(long)(n0 + ty + i * 8) * K + k0 + tx] = (__bf16)tile[tx][ty + i * 8];
}

// ---------------------------------------------------------------------------
// Per-head transpose of V: V[b][s][h*HD+hd] -> Vt[(b*H+h)][hd][s]  (bf16).
// Makes every attention V-tile a clean strided 2D tile for the TDM.
// Grid: (S/32, HD/32, B*H); block 256 (32x8 logical).
// ---------------------------------------------------------------------------
__global__ __launch_bounds__(256)
void transpose_v_bf16(const __bf16* __restrict__ V, __bf16* __restrict__ Vt) {
  __shared__ __bf16 tile[32][33];
  const int tx  = threadIdx.x & 31;
  const int ty  = threadIdx.x >> 5;
  const int s0  = blockIdx.x * 32;
  const int hd0 = blockIdx.y * 32;
  const int bh  = blockIdx.z;                       // b*H + h
  const int b   = bh >> 5, h = bh & 31;
  const __bf16* src = V + ((long)b * cfg::S) * cfg::D + (long)h * cfg::HD;
#pragma unroll
  for (int i = 0; i < 4; ++i)
    tile[ty + i * 8][tx] = src[(long)(s0 + ty + i * 8) * cfg::D + hd0 + tx];
  __syncthreads();
  __bf16* dst = Vt + (long)bh * cfg::HD * cfg::S;
#pragma unroll
  for (int i = 0; i < 4; ++i)
    dst[(long)(hd0 + ty + i * 8) * cfg::S + s0 + tx] = tile[tx][ty + i * 8];
}

// ---------------------------------------------------------------------------
// bf16 GEMM with fp32 accumulation:
//   C[M][N] = A[M][K] * Bt[N][K]^T        (Bt is pre-transposed weights)
// Block tile 128x128, 8 waves (2x4), wave tile 64x32 = 4x2 WMMA tiles.
// K staged in LDS 64 deep via double-buffered TDM DMA issued by wave 0:
// DMA of slab k+1 overlaps WMMA on slab k.
// ---------------------------------------------------------------------------
constexpr int GBM = 128, GBN = 128, GBK = 64;
constexpr int GLD = GBK + 8;   // 72 bf16 -> 144B row stride (TDM pad gives this)

template <typename OutT>
__global__ __launch_bounds__(256)
void gemm_bf16_wmma(const __bf16* __restrict__ A, const __bf16* __restrict__ Bt,
                    OutT* __restrict__ C, int Mdim, int Ndim, int Kdim) {
  __shared__ __bf16 As[2][GBM * GLD];
  __shared__ __bf16 Bs[2][GBM * GLD];

  const int tid  = threadIdx.x;
  const int wave = tid >> 5;
  const int lane = tid & 31;
  const int half = lane >> 4;
  const int l15  = lane & 15;
  const int wm   = (wave >> 2) * 64;   // 0 or 64
  const int wn   = (wave & 3) * 32;    // 0,32,64,96
  const int bm0  = blockIdx.y * GBM;
  const int bn0  = blockIdx.x * GBN;

  // Scalar-uniform wave-0 test so the compiler emits an s_cbranch: TDM ops
  // ignore EXEC, so an EXEC-masked region would issue duplicate DMAs.
  const bool tdm_wave = (__builtin_amdgcn_readfirstlane((int)threadIdx.x) < 32);

  f32x8 acc[4][2];
#pragma unroll
  for (int i = 0; i < 4; ++i)
#pragma unroll
    for (int j = 0; j < 2; ++j)
      acc[i][j] = f32x8{0.f, 0.f, 0.f, 0.f, 0.f, 0.f, 0.f, 0.f};

  // Prime buffer 0.  Tile 128 rows x 64 bf16, row stride 4096;
  // pad: 32 DW interval (code 4), 4 DW pad (code 3) -> 144B LDS rows.
  if (tdm_wave) {
    tdm_load_2d(A  + (long)bm0 * Kdim, lds_byte_off(&As[0][0]),
                64, 128, 4096, 4096, 4096, 4, 3);
    tdm_load_2d(Bt + (long)bn0 * Kdim, lds_byte_off(&Bs[0][0]),
                64, 128, 4096, 4096, 4096, 4, 3);
  }

  int buf = 0;
  for (int kk = 0; kk < Kdim; kk += GBK) {
    if (tdm_wave) __builtin_amdgcn_s_wait_tensorcnt(0);
    __syncthreads();   // current buffer DMA complete + prev reads of other buf done

    if (tdm_wave && (kk + GBK < Kdim)) {
      tdm_load_2d(A  + (long)bm0 * Kdim + kk + GBK,
                  lds_byte_off(&As[buf ^ 1][0]),
                  64, 128, 4096, 4096, 4096, 4, 3);
      tdm_load_2d(Bt + (long)bn0 * Kdim + kk + GBK,
                  lds_byte_off(&Bs[buf ^ 1][0]),
                  64, 128, 4096, 4096, 4096, 4, 3);
    }

    const __bf16* as = As[buf];
    const __bf16* bs = Bs[buf];
#pragma unroll
    for (int kc = 0; kc < GBK; kc += 32) {
      bf16x16 af[4], bfg[2];
#pragma unroll
      for (int s = 0; s < 4; ++s) {
        const __bf16* p = &as[(wm + s * 16 + l15) * GLD + kc + half * 8];
        af[s] = ld_frag16(p, p + 16);
      }
#pragma unroll
      for (int s = 0; s < 2; ++s) {
        const __bf16* p = &bs[(wn + s * 16 + l15) * GLD + kc + half * 8];
        bfg[s] = ld_frag16(p, p + 16);
      }
#pragma unroll
      for (int i = 0; i < 4; ++i)
#pragma unroll
        for (int j = 0; j < 2; ++j)
          acc[i][j] = wmma_bf16(af[i], bfg[j], acc[i][j]);
    }
    buf ^= 1;
  }

  // C/D layout: VGPR e -> row e + 8*half, col = l15 (within 16x16 subtile)
#pragma unroll
  for (int i = 0; i < 4; ++i)
#pragma unroll
    for (int j = 0; j < 2; ++j)
#pragma unroll
      for (int e = 0; e < 8; ++e) {
        int row = bm0 + wm + i * 16 + e + half * 8;
        int col = bn0 + wn + j * 16 + l15;
        C[(long)row * Ndim + col] = (OutT)acc[i][j][e];
      }
}

// ---------------------------------------------------------------------------
// RoPE applied in-place to Q and K (bf16 storage, fp32 math).
// One thread per (token, head, pair).
// ---------------------------------------------------------------------------
__global__ __launch_bounds__(256)
void rope_kernel(__bf16* __restrict__ Q, __bf16* __restrict__ K) {
  const long idx  = (long)blockIdx.x * blockDim.x + threadIdx.x;
  const int  pair = (int)(idx & 63);            // HD/2 = 64
  const long r1   = idx >> 6;
  const int  h    = (int)(r1 & 31);             // H = 32
  const long tok  = r1 >> 5;                    // 0 .. B*S-1
  const int  spos = (int)(tok & (cfg::S - 1));  // position within sequence

  // inv_freq = 10000^(-2*pair/128)
  const float inv = __expf((float)pair * (-2.0f / (float)cfg::HD) * 9.210340372f);
  const float ang = (float)spos * inv;
  float sn, cs;
  __sincosf(ang, &sn, &cs);

  const long base = tok * cfg::D + (long)h * cfg::HD + 2 * pair;
  {
    float x0 = (float)Q[base], x1 = (float)Q[base + 1];
    Q[base]     = (__bf16)(x0 * cs - x1 * sn);
    Q[base + 1] = (__bf16)(x0 * sn + x1 * cs);
  }
  {
    float x0 = (float)K[base], x1 = (float)K[base + 1];
    K[base]     = (__bf16)(x0 * cs - x1 * sn);
    K[base + 1] = (__bf16)(x0 * sn + x1 * cs);
  }
}

// ---------------------------------------------------------------------------
// Causal flash attention, bf16 WMMA, fp32 accumulation + online softmax.
// Grid: (S/64, H, B). Block: 128 threads = 4 waves; wave w owns q rows
// [qb*64 + w*16, +16). K tiles [32 keys][128 hd] and pre-transposed V tiles
// [128 hd][32 keys] are DMA'd by the TDM, double-buffered: wave 0 issues the
// DMA for tile t+1 while all waves compute tile t. No vector staging code.
// ---------------------------------------------------------------------------
constexpr int FA_KT  = 32;
constexpr int FA_KLD = cfg::HD + 8;   // 136 -> 272B stride (TDM pad 5/3)
constexpr int FA_VLD = FA_KT + 8;     // 40  -> 80B stride  (TDM pad 3/3)

__global__ __launch_bounds__(128)
void fa_kernel(const __bf16* __restrict__ Q, const __bf16* __restrict__ K,
               const __bf16* __restrict__ Vt, __bf16* __restrict__ O) {
  __shared__ __bf16 Ksm[2][FA_KT * FA_KLD];     // [key][hd]
  __shared__ __bf16 Vtsm[2][cfg::HD * FA_VLD];  // [hd][key]
  __shared__ __bf16 Psm[64 * FA_VLD];           // [wave*16 + qrow][key]

  const int qb   = blockIdx.x;
  const int h    = blockIdx.y;
  const int b    = blockIdx.z;
  const int tid  = threadIdx.x;
  const int wave = tid >> 5;
  const int lane = tid & 31;
  const int half = lane >> 4;
  const int l15  = lane & 15;
  const int q0   = qb * 64 + wave * 16;

  const bool tdm_wave = (__builtin_amdgcn_readfirstlane(tid) < 32);

  // Q A-fragments straight from global: lane row = l15, contiguous hd chunks.
  bf16x16 qf[4];
  {
    const __bf16* qp = Q + ((long)b * cfg::S + q0 + l15) * cfg::D + (long)h * cfg::HD;
#pragma unroll
    for (int c = 0; c < 4; ++c) {
      const __bf16* p = qp + c * 32 + half * 8;
      qf[c] = ld_frag16(p, p + 16);
    }
  }

  // K tile source: rows = tokens (stride D), 128 bf16 segment at h*HD.
  const __bf16* kbase = K + ((long)b * cfg::S) * cfg::D + (long)h * cfg::HD;
  // V^T tile source: rows = hd (stride S), 32-key segment at column t*32.
  const __bf16* vtbase = Vt + (long)(b * cfg::H + h) * cfg::HD * cfg::S;

  f32x8 o[8];
#pragma unroll
  for (int t = 0; t < 8; ++t)
    o[t] = f32x8{0.f, 0.f, 0.f, 0.f, 0.f, 0.f, 0.f, 0.f};
  float mrow[8], lrow[8];
#pragma unroll
  for (int j = 0; j < 8; ++j) { mrow[j] = -3.0e38f; lrow[j] = 0.0f; }

  const int ntiles = 2 * qb + 2;   // keys up to qb*64+63

  if (tdm_wave) {
    tdm_load_2d(kbase, lds_byte_off(&Ksm[0][0]),
                128, 32, 4096, 4096, 4096, 5, 3);   // 64 DW rows + 16B pad
    tdm_load_2d(vtbase, lds_byte_off(&Vtsm[0][0]),
                32, 128, 2048, 2048, 128, 3, 3);    // 16 DW rows + 16B pad
  }

  int bufi = 0;
  for (int t = 0; t < ntiles; ++t) {
    if (tdm_wave) __builtin_amdgcn_s_wait_tensorcnt(0);
    __syncthreads();   // tile t DMA done; prior reads of other buffer done

    if (tdm_wave && (t + 1 < ntiles)) {
      tdm_load_2d(kbase + (long)(t + 1) * FA_KT * cfg::D,
                  lds_byte_off(&Ksm[bufi ^ 1][0]),
                  128, 32, 4096, 4096, 4096, 5, 3);
      tdm_load_2d(vtbase + (long)(t + 1) * FA_KT,
                  lds_byte_off(&Vtsm[bufi ^ 1][0]),
                  32, 128, 2048, 2048, 128, 3, 3);
    }

    const __bf16* ks = Ksm[bufi];
    const __bf16* vs = Vtsm[bufi];

    // S = (Q . K^T) * scale, with causal mask, as two 16x16 C-fragments.
    f32x8 sfrag[2];
#pragma unroll
    for (int ns = 0; ns < 2; ++ns) {
      f32x8 s = f32x8{0.f, 0.f, 0.f, 0.f, 0.f, 0.f, 0.f, 0.f};
#pragma unroll
      for (int c = 0; c < 4; ++c) {
        const __bf16* p = &ks[(ns * 16 + l15) * FA_KLD + c * 32 + half * 8];
        s = wmma_bf16(qf[c], ld_frag16(p, p + 16), s);
      }
      const int col = t * FA_KT + ns * 16 + l15;
#pragma unroll
      for (int j = 0; j < 8; ++j) {
        const int row = q0 + j + half * 8;
        float v = s[j] * cfg::SM_SCALE;
        s[j] = (col <= row) ? v : -1.0e30f;
      }
      sfrag[ns] = s;
    }

    // Online softmax per row (rows split: VGPR j + 8*half; cols across 16-lane
    // halves, so xor-shuffles 8/4/2/1 reduce within the correct half).
#pragma unroll
    for (int j = 0; j < 8; ++j) {
      float mx = fmaxf(sfrag[0][j], sfrag[1][j]);
#pragma unroll
      for (int msk = 8; msk >= 1; msk >>= 1)
        mx = fmaxf(mx, __shfl_xor(mx, msk, 32));
      const float mnew  = fmaxf(mrow[j], mx);
      const float alpha = __expf(mrow[j] - mnew);
      const float p0 = __expf(sfrag[0][j] - mnew);
      const float p1 = __expf(sfrag[1][j] - mnew);
      sfrag[0][j] = p0;
      sfrag[1][j] = p1;
      float rs = p0 + p1;
#pragma unroll
      for (int msk = 8; msk >= 1; msk >>= 1)
        rs += __shfl_xor(rs, msk, 32);
      lrow[j] = lrow[j] * alpha + rs;
      mrow[j] = mnew;
#pragma unroll
      for (int nt = 0; nt < 8; ++nt) o[nt][j] *= alpha;
    }

    // C-layout -> A-layout for P via per-wave LDS region (same-wave LDS ops
    // are in-order; dependency waits inserted by compiler).
#pragma unroll
    for (int ns = 0; ns < 2; ++ns)
#pragma unroll
      for (int j = 0; j < 8; ++j)
        Psm[(wave * 16 + j + half * 8) * FA_VLD + ns * 16 + l15] =
            (__bf16)sfrag[ns][j];

    const __bf16* pp = &Psm[(wave * 16 + l15) * FA_VLD + half * 8];
    const bf16x16 pf = ld_frag16(pp, pp + 16);

    // O += P . V  across 8 hd subtiles; B-frag from transposed V tile.
#pragma unroll
    for (int nt = 0; nt < 8; ++nt) {
      const __bf16* vp = &vs[(nt * 16 + l15) * FA_VLD + half * 8];
      o[nt] = wmma_bf16(pf, ld_frag16(vp, vp + 16), o[nt]);
    }

    bufi ^= 1;
  }

  // Epilogue: O /= l, write bf16 [b][s][h*HD+hd].
#pragma unroll
  for (int j = 0; j < 8; ++j) {
    const float invl = 1.0f / lrow[j];
    const int   row  = q0 + j + half * 8;
    const long  base = ((long)b * cfg::S + row) * cfg::D + (long)h * cfg::HD;
#pragma unroll
    for (int nt = 0; nt < 8; ++nt)
      O[base + nt * 16 + l15] = (__bf16)(o[nt][j] * invl);
  }
}

// ---------------------------------------------------------------------------
// Host launcher
// ---------------------------------------------------------------------------
extern "C" void kernel_launch(void* const* d_in, const int* in_sizes, int n_in,
                              void* d_out, int out_size, void* d_ws, size_t ws_size,
                              hipStream_t stream) {
  (void)in_sizes; (void)n_in; (void)out_size; (void)ws_size;
  const float* x  = (const float*)d_in[0];
  const float* wq = (const float*)d_in[1];
  const float* wk = (const float*)d_in[2];
  const float* wv = (const float*)d_in[3];
  const float* wo = (const float*)d_in[4];
  float* out = (float*)d_out;

  char* ws = (char*)d_ws;
  const size_t SZ = (size_t)cfg::M * cfg::D * sizeof(__bf16);  // 32 MB
  __bf16* Xb = (__bf16*)(ws);            // x bf16; reused as attention output
  __bf16* Wt = (__bf16*)(ws + 1 * SZ);   // transposed weight (reused 4x)
  __bf16* Qb = (__bf16*)(ws + 2 * SZ);
  __bf16* Kb = (__bf16*)(ws + 3 * SZ);
  __bf16* Vb = (__bf16*)(ws + 4 * SZ);
  __bf16* Vt = (__bf16*)(ws + 5 * SZ);   // per-head transposed V
  __bf16* AO = Xb;                       // attention output (Xb dead by then)

  const long nx = (long)cfg::M * cfg::D;
  convert_f32_to_bf16<<<4096, 256, 0, stream>>>(x, Xb, nx);

  const dim3 tgrid(cfg::D / 32, cfg::D / 32), tblk(256);
  const dim3 ggrid(cfg::D / GBN, cfg::M / GBM), gblk(256);

  // Q = X Wq
  transpose_convert_bf16<<<tgrid, tblk, 0, stream>>>(wq, Wt, cfg::D, cfg::D);
  gemm_bf16_wmma<__bf16><<<ggrid, gblk, 0, stream>>>(Xb, Wt, Qb, cfg::M, cfg::D, cfg::D);
  // K = X Wk
  transpose_convert_bf16<<<tgrid, tblk, 0, stream>>>(wk, Wt, cfg::D, cfg::D);
  gemm_bf16_wmma<__bf16><<<ggrid, gblk, 0, stream>>>(Xb, Wt, Kb, cfg::M, cfg::D, cfg::D);
  // V = X Wv
  transpose_convert_bf16<<<tgrid, tblk, 0, stream>>>(wv, Wt, cfg::D, cfg::D);
  gemm_bf16_wmma<__bf16><<<ggrid, gblk, 0, stream>>>(Xb, Wt, Vb, cfg::M, cfg::D, cfg::D);

  // RoPE(Q, K)
  const long nrope = (long)cfg::B * cfg::S * cfg::H * (cfg::HD / 2);
  rope_kernel<<<(unsigned)(nrope / 256), 256, 0, stream>>>(Qb, Kb);

  // V^T per head for TDM-friendly attention tiles
  transpose_v_bf16<<<dim3(cfg::S / 32, cfg::HD / 32, cfg::B * cfg::H), 256, 0, stream>>>(Vb, Vt);

  // Attention -> AO (overwrites Xb, which is no longer needed)
  fa_kernel<<<dim3(cfg::S / 64, cfg::H, cfg::B), 128, 0, stream>>>(Qb, Kb, Vt, AO);

  // out = AO Wo (fp32 output)
  transpose_convert_bf16<<<tgrid, tblk, 0, stream>>>(wo, Wt, cfg::D, cfg::D);
  gemm_bf16_wmma<float><<<ggrid, gblk, 0, stream>>>(AO, Wt, out, cfg::M, cfg::D, cfg::D);
}
